// GCN_generator_6459630814174
// MI455X (gfx1250) — matrix operand order repaired and checked by reading, not compiled, not run on hardware
//
#include <hip/hip_runtime.h>
#include <hip/hip_bf16.h>

typedef __attribute__((ext_vector_type(16))) __bf16 v16bf;
typedef __attribute__((ext_vector_type(8)))  float  v8f;
typedef __attribute__((ext_vector_type(4)))  __bf16 v4bf;

#define Bn 4
#define Nn 256
#define Dn 128
#define NTHREADS 256
#define NWAVES (NTHREADS / 32)

// LDS layout (bytes):
//   xTb : [Dn][Nn] bf16  = 65536   (x transposed; B-fragments for adj@x)
//   xTf : [Dn][Nn] f32   = 131072  (f32 master copy of x; prob dots)
//   yb  : [Nn][Dn] bf16  = 65536   (y = adj_n @ x, row-major; A of y@W)
//   WT  : [Dn][Dn] bf16  = 32768   (W transposed; B-fragments for y@W)
//   dinv: [Nn]     f32   = 1024
#define SM_XTB  0
#define SM_XTF  (SM_XTB + Dn * Nn * 2)
#define SM_YB   (SM_XTF + Dn * Nn * 4)
#define SM_WT   (SM_YB + Nn * Dn * 2)
#define SM_DINV (SM_WT + Dn * Dn * 2)
#define SM_SIZE (SM_DINV + Nn * 4)

// A-fragment (16-bit, 16x32): lane m = m0 + lane%16; K elems 0..7 at kbase,
// elems 8..15 at kbase+16, where kbase = k0 + (lane/16)*8.
__device__ __forceinline__ v16bf load_fragA(const __bf16* base, int stride,
                                            int m, int kbase) {
  union { v16bf v; int4 q[2]; } u;
  const __bf16* p = base + m * stride + kbase;
  u.q[0] = *(const int4*)(p);
  u.q[1] = *(const int4*)(p + 16);
  return u.v;
}

// B-fragment (16-bit, 32x16): lane n = n0 + lane%16; K elems 0..7 at kbase,
// elems 8..15 at kbase+16, where kbase = k0 + (lane/16)*8 (K-major transpose in memory).
__device__ __forceinline__ v16bf load_fragB(const __bf16* base, int stride,
                                            int n, int kbase) {
  union { v16bf v; int4 q[2]; } u;
  const __bf16* p = base + n * stride + kbase;
  u.q[0] = *(const int4*)(p);
  u.q[1] = *(const int4*)(p + 16);
  return u.v;
}

// x_new = relu(yb[256x128] @ W) ; writes xTb (bf16) and xTf (f32), transposed.
__device__ __forceinline__ void matmul_yW(const __bf16* yb, const __bf16* WT,
                                          __bf16* xTb, float* xTf,
                                          int wave, int lane) {
  const int col = lane & 15;
  const int half = lane >> 4;
  for (int rbp = 0; rbp < 2; ++rbp) {
    const int m0 = (wave + rbp * NWAVES) * 16;
    v8f acc[8];
    const v8f zero = {0.f, 0.f, 0.f, 0.f, 0.f, 0.f, 0.f, 0.f};
#pragma unroll
    for (int t = 0; t < 8; ++t) acc[t] = zero;
    for (int k0 = 0; k0 < Dn; k0 += 32) {
      v16bf a = load_fragA(yb, Dn, m0 + col, k0 + half * 8);
#pragma unroll
      for (int nt = 0; nt < 8; ++nt) {
        v16bf bb = load_fragB(WT, Dn, nt * 16 + col, k0 + half * 8);
        acc[nt] = __builtin_amdgcn_wmma_f32_16x16x32_bf16(
            false, a, false, bb, (short)0, acc[nt], false, false);
      }
    }
    const int mrow = m0 + half * 8;
#pragma unroll
    for (int nt = 0; nt < 8; ++nt) {
#pragma unroll
      for (int r = 0; r < 8; ++r) {
        float v = fmaxf(acc[nt][r], 0.0f);
        int d = nt * 16 + col;
        int m = mrow + r;
        xTb[d * Nn + m] = (__bf16)v;
        xTf[d * Nn + m] = v;
      }
    }
  }
}

__global__ void __launch_bounds__(NTHREADS)
gcn_generator_kernel(const float* __restrict__ x_in,
                     const float* __restrict__ W_in,
                     float* __restrict__ out,
                     float* __restrict__ adjf_ws,
                     void* __restrict__ adjb_ws) {
  extern __shared__ char smem[];
  __bf16* xTb = (__bf16*)(smem + SM_XTB);
  float*  xTf = (float*)(smem + SM_XTF);
  __bf16* yb  = (__bf16*)(smem + SM_YB);
  __bf16* WT  = (__bf16*)(smem + SM_WT);
  float*  dinv = (float*)(smem + SM_DINV);

  const int b = blockIdx.x;
  const int tid = threadIdx.x;
  const int lane = tid & 31;
  const int wave = tid >> 5;
  const int col = lane & 15;
  const int half = lane >> 4;

  const float* xg = x_in + (size_t)b * Nn * Dn;
  float* adjf = adjf_ws + (size_t)b * Nn * Nn;
  __bf16* adjb = (__bf16*)adjb_ws + (size_t)b * Nn * Nn;
  float* outb = out + (size_t)b * Nn * Nn;

  // ---------- init: adj = adj_out = I ; stage x (bf16) and W^T ----------
  for (int idx = tid; idx < Nn * Nn; idx += NTHREADS) {
    float v = ((idx >> 8) == (idx & 255)) ? 1.0f : 0.0f;
    adjf[idx] = v;
    outb[idx] = v;
    adjb[idx] = (__bf16)v;
  }
  for (int idx = tid; idx < Nn * Dn; idx += NTHREADS)
    yb[idx] = (__bf16)xg[idx];  // x row-major bf16, reuse yb as A buffer
  for (int idx = tid; idx < Dn * Dn; idx += NTHREADS) {
    int k = idx >> 7, n = idx & 127;
    WT[n * Dn + k] = (__bf16)W_in[idx];
  }
  __threadfence();
  __syncthreads();

  // x = relu(x @ W) -> xTb / xTf
  matmul_yW(yb, WT, xTb, xTf, wave, lane);
  __syncthreads();

  // ---------- sequential scan over i ----------
  for (int i = 1; i < Nn; ++i) {
    // phase 1: prob[j] = <x[j], x[i]>; scatter into adj and adj_out (j < i)
    {
      const int j = tid;
      float s = 0.f;
#pragma unroll 4
      for (int d = 0; d < Dn; ++d)
        s = fmaf(xTf[d * Nn + j], xTf[d * Nn + i], s);
      if (j < i) {
        adjf[i * Nn + j] = s;
        adjf[j * Nn + i] = s;
        outb[i * Nn + j] = s;
        outb[j * Nn + i] = s;
      }
    }
    __threadfence();
    __syncthreads();

    // phase 2a: deg -> dinv = rsqrt(deg) (thread tid owns row tid)
    {
      const float4* row = (const float4*)(adjf + tid * Nn);
      float s = 0.f;
      for (int q = 0; q < Nn / 4; ++q) {
        float4 v = row[q];
        s += v.x + v.y + v.z + v.w;
      }
      dinv[tid] = rsqrtf(s);
    }
    __syncthreads();

    // phase 2b: adj_n = dinv_i * adj * dinv_j (f32 master + bf16 shadow)
    for (int q = tid; q < (Nn * Nn) / 4; q += NTHREADS) {
      int e = q << 2;
      int n = e >> 8;
      int j0 = e & 255;
      float dn = dinv[n];
      float4 v = ((const float4*)adjf)[q];
      v.x *= dn * dinv[j0 + 0];
      v.y *= dn * dinv[j0 + 1];
      v.z *= dn * dinv[j0 + 2];
      v.w *= dn * dinv[j0 + 3];
      ((float4*)adjf)[q] = v;
      v4bf pb;
      pb[0] = (__bf16)v.x; pb[1] = (__bf16)v.y;
      pb[2] = (__bf16)v.z; pb[3] = (__bf16)v.w;
      *(v4bf*)(adjb + e) = pb;
    }
    __threadfence();
    __syncthreads();

    // phase 3: y = adj_n @ x  (A = adjb in L2, B = xTb in LDS) -> yb
    for (int rbp = 0; rbp < 2; ++rbp) {
      const int m0 = (wave + rbp * NWAVES) * 16;
      v8f acc[8];
      const v8f zero = {0.f, 0.f, 0.f, 0.f, 0.f, 0.f, 0.f, 0.f};
#pragma unroll
      for (int t = 0; t < 8; ++t) acc[t] = zero;
      for (int k0 = 0; k0 < Nn; k0 += 32) {
        v16bf a = load_fragA(adjb, Nn, m0 + col, k0 + half * 8);
        if (k0 + 32 < Nn)
          __builtin_prefetch(adjb + (m0 + col) * Nn + k0 + 32, 0, 0);
#pragma unroll
        for (int nt = 0; nt < 8; ++nt) {
          v16bf bb = load_fragB(xTb, Nn, nt * 16 + col, k0 + half * 8);
          acc[nt] = __builtin_amdgcn_wmma_f32_16x16x32_bf16(
              false, a, false, bb, (short)0, acc[nt], false, false);
        }
      }
      const int mrow = m0 + half * 8;
#pragma unroll
      for (int nt = 0; nt < 8; ++nt) {
#pragma unroll
        for (int r = 0; r < 8; ++r)
          yb[(mrow + r) * Dn + nt * 16 + col] = (__bf16)acc[nt][r];
      }
    }
    // all waves must finish reading xTb (phase-3 B operand) before phase 4
    // overwrites it; each wave reads only its own yb rows, so one barrier.
    __syncthreads();

    // phase 4: x = relu(y @ W) -> xTb / xTf
    matmul_yW(yb, WT, xTb, xTf, wave, lane);
    __syncthreads();
  }
}

extern "C" void kernel_launch(void* const* d_in, const int* in_sizes, int n_in,
                              void* d_out, int out_size, void* d_ws,
                              size_t ws_size, hipStream_t stream) {
  const float* x = (const float*)d_in[0];  // (B, N, D) f32
  const float* W = (const float*)d_in[1];  // (D, D)    f32
  float* out = (float*)d_out;              // (B, N, N) f32

  // workspace: f32 adjacency master + bf16 shadow for WMMA fragments
  float* adjf = (float*)d_ws;
  void* adjb = (void*)((char*)d_ws + (size_t)Bn * Nn * Nn * sizeof(float));

  (void)in_sizes; (void)n_in; (void)out_size; (void)ws_size;

  (void)hipFuncSetAttribute((const void*)gcn_generator_kernel,
                            hipFuncAttributeMaxDynamicSharedMemorySize,
                            (int)SM_SIZE);
  gcn_generator_kernel<<<dim3(Bn), dim3(NTHREADS), SM_SIZE, stream>>>(
      x, W, out, adjf, adjb);
}
